// QWen3MoeAttention_24618752541337
// MI455X (gfx1250) — matrix operand order repaired and checked by reading, not compiled, not run on hardware
//
#include <hip/hip_runtime.h>
#include <hip/hip_bf16.h>

// ---------------------------------------------------------------------------
// Types for CDNA5 WMMA (wave32): v_wmma_f32_16x16x32_bf16
// ---------------------------------------------------------------------------
typedef __bf16 bf16_t;
typedef __attribute__((ext_vector_type(16))) __bf16 v16bf;
typedef __attribute__((ext_vector_type(8)))  __bf16 v8bf;   // 16-byte vector
typedef __attribute__((ext_vector_type(8)))  float  v8f;
typedef __attribute__((ext_vector_type(4)))  unsigned int uint32x4;
typedef __attribute__((ext_vector_type(8)))  int int32x8;
typedef __attribute__((ext_vector_type(4)))  int int32x4;

#define T_SEQ     2048
#define HIDDEN    2048
#define NUM_HEADS 32
#define NUM_KVH   4
#define HEAD_DIM  128
#define QDIM      (NUM_HEADS * HEAD_DIM)   // 4096
#define KVDIM     (NUM_KVH * HEAD_DIM)     // 512

__device__ inline v8f v8f_zero() {
  v8f z = {0.f, 0.f, 0.f, 0.f, 0.f, 0.f, 0.f, 0.f};
  return z;
}

__device__ inline v8f wmma_bf16(v16bf a, v16bf b, v8f c) {
  return __builtin_amdgcn_wmma_f32_16x16x32_bf16(false, a, false, b, (short)0, c,
                                                 false, false);
}

// ---------------------------------------------------------------------------
// Tensor Data Mover: async 2D row-major tile DMA global -> LDS (6-arg builtin).
// D# groups 0/1 per CDNA5 ISA 8.3/8.4; LDS padding writes directly into a
// padded [rows][row+pad] layout. Issue from ONE wave; that wave must
// s_wait_tensorcnt(0) before the releasing barrier.
// ---------------------------------------------------------------------------
__device__ inline void tdm_load_2d(const void* gaddr, unsigned lds_off,
                                   int tensor_d0, int tensor_d1,
                                   int tile_d0, int tile_d1,
                                   int pad_icode, int pad_acode) {
  unsigned long long ga = (unsigned long long)gaddr;
  uint32x4 g0;
  g0[0] = 1u;                                        // count=1, user mode
  g0[1] = lds_off;                                   // LDS byte address
  g0[2] = (unsigned)(ga & 0xffffffffu);              // global_addr[31:0]
  g0[3] = (unsigned)((ga >> 32) & 0x1ffffffu) | (2u << 30);  // addr[56:32], type=2
  int32x8 g1;
  unsigned w0 = (1u << 16)                                 // data_size = 2 bytes
              | (1u << 20)                                 // pad_enable
              | ((unsigned)pad_icode << 22)
              | ((unsigned)pad_acode << 25);
  g1[0] = (int)w0;
  g1[1] = (int)(((unsigned)tensor_d0 & 0xffffu) << 16);    // tensor_dim0 lo16
  g1[2] = (int)((((unsigned)tensor_d0 >> 16) & 0xffffu) |
                (((unsigned)tensor_d1 & 0xffffu) << 16));  // d0 hi16 | d1 lo16
  g1[3] = (int)((((unsigned)tensor_d1 >> 16) & 0xffffu) |
                ((unsigned)tile_d0 << 16));                // d1 hi16 | tile_dim0
  g1[4] = (int)((unsigned)tile_d1 & 0xffffu);              // tile_dim1 (tile_dim2=0)
  g1[5] = tensor_d0;                                       // tensor_dim0_stride lo32
  g1[6] = 0;
  g1[7] = 0;
  int32x4 gz4 = {0, 0, 0, 0};
  int32x8 gz8 = {0, 0, 0, 0, 0, 0, 0, 0};
  __builtin_amdgcn_tensor_load_to_lds(g0, g1, gz4, gz4, gz8, 0);
}

__device__ inline unsigned lds_offset_of(const void* p) {
  // Flat LDS addresses carry the in-LDS byte offset in addr[31:0] (ISA 10.2).
  return (unsigned)(unsigned long long)p;
}

// A-matrix fragment (16-bit, 16x32) from row-major LDS tile [m][k].
__device__ inline v16bf load_frag_a(const bf16_t* base, int ld, int lane) {
  int m = lane & 15, half = lane >> 4;
  const bf16_t* p = base + m * ld + half * 16;
  v16bf a;
#pragma unroll
  for (int i = 0; i < 8; ++i) {
    a[i]     = p[i];
    a[8 + i] = p[8 + i];
  }
  return a;
}

// B-matrix fragment (16-bit, 32x16) from transposed LDS tile [n][k].
__device__ inline v16bf load_frag_bt(const bf16_t* base, int ld, int lane) {
  int n = lane & 15, half = lane >> 4;
  const bf16_t* p = base + n * ld + half * 16;
  v16bf b;
#pragma unroll
  for (int i = 0; i < 16; ++i) b[i] = p[i];
  return b;
}

// ---------------------------------------------------------------------------
// fp32 -> bf16 convert (grid-stride)
// ---------------------------------------------------------------------------
__global__ void f32_to_bf16_kernel(const float* __restrict__ in,
                                   bf16_t* __restrict__ out, long n) {
  long i = (long)blockIdx.x * blockDim.x + threadIdx.x;
  long stride = (long)gridDim.x * blockDim.x;
  for (; i < n; i += stride) out[i] = (bf16_t)in[i];
}

// ---------------------------------------------------------------------------
// bf16 GEMM: C[2048xN] = A[2048xK] @ B[KxN], row-major; N/K compile-time.
// 256 threads = 8 waves; 128x128 tile/WG; wave: 32x64 (2x4 WMMA tiles).
// Double-buffered: TDM stages A tile i+1 and registers hold B tile i+1 while
// WMMAs run on tile i. One barrier per k-step.
// ---------------------------------------------------------------------------
template <int N, int K, bool OUT_BF16>
__global__ __launch_bounds__(256)
void gemm_bf16_wmma(const bf16_t* __restrict__ A, const bf16_t* __restrict__ B,
                    void* __restrict__ Cv) {
  __shared__ bf16_t As[2][128 * 40];   // [m][k] padded (TDM pad: 16DW + 4DW)
  __shared__ bf16_t Bs[2][128 * 40];   // [n][k] transposed tile

  const int tid  = threadIdx.x;
  const int lane = tid & 31;
  const int wave = tid >> 5;
  const int wrow = wave >> 1;
  const int wcol = wave & 1;
  const int bm = blockIdx.y * 128;
  const int bn = blockIdx.x * 128;

  v8f acc[2][4];
#pragma unroll
  for (int i = 0; i < 2; ++i)
#pragma unroll
    for (int j = 0; j < 4; ++j) acc[i][j] = v8f_zero();

  // ---- prologue: stage tile 0 into buffer 0
  {
    if (wave == 0)
      tdm_load_2d(A + (long)bm * K, lds_offset_of(&As[0][0]),
                  K, 2048, 32, 128, /*pad_icode=*/3, /*pad_acode=*/3);
    v8bf bv[2];
#pragma unroll
    for (int it = 0; it < 2; ++it) {
      int u = tid + it * 256;
      int oc = (u >> 5) << 3, row = u & 31;
      bv[it] = *(const v8bf*)(B + (long)row * N + bn + oc);
    }
#pragma unroll
    for (int it = 0; it < 2; ++it) {
      int u = tid + it * 256;
      int oc = (u >> 5) << 3, row = u & 31;
#pragma unroll
      for (int e = 0; e < 8; ++e) Bs[0][(oc + e) * 40 + row] = bv[it][e];
    }
    if (wave == 0) __builtin_amdgcn_s_wait_tensorcnt(0);
    __syncthreads();
  }

  int buf = 0;
  for (int k0 = 0; k0 < K; k0 += 32, buf ^= 1) {
    const int nk = k0 + 32;
    const bool pref = (nk < K);
    // ---- prefetch tile i+1 (TDM A -> LDS, B -> registers)
    v8bf bv[2];
    if (pref) {
      if (wave == 0)
        tdm_load_2d(A + (long)bm * K + nk, lds_offset_of(&As[buf ^ 1][0]),
                    K, 2048, 32, 128, 3, 3);
#pragma unroll
      for (int it = 0; it < 2; ++it) {
        int u = tid + it * 256;
        int oc = (u >> 5) << 3, row = u & 31;
        bv[it] = *(const v8bf*)(B + (long)(nk + row) * N + bn + oc);
      }
    }
    // ---- compute on tile i: preload all fragments, then WMMA chain
    v16bf af0 = load_frag_a(&As[buf][(wrow * 32 + 0)  * 40], 40, lane);
    v16bf af1 = load_frag_a(&As[buf][(wrow * 32 + 16) * 40], 40, lane);
    v16bf bfr[4];
#pragma unroll
    for (int j = 0; j < 4; ++j)
      bfr[j] = load_frag_bt(&Bs[buf][(wcol * 64 + j * 16) * 40], 40, lane);
#pragma unroll
    for (int j = 0; j < 4; ++j) {
      acc[0][j] = wmma_bf16(af0, bfr[j], acc[0][j]);
      acc[1][j] = wmma_bf16(af1, bfr[j], acc[1][j]);
    }
    // ---- commit tile i+1, release
    if (pref) {
#pragma unroll
      for (int it = 0; it < 2; ++it) {
        int u = tid + it * 256;
        int oc = (u >> 5) << 3, row = u & 31;
#pragma unroll
        for (int e = 0; e < 8; ++e) Bs[buf ^ 1][(oc + e) * 40 + row] = bv[it][e];
      }
      if (wave == 0) __builtin_amdgcn_s_wait_tensorcnt(0);
    }
    __syncthreads();
  }

  const int n = lane & 15, half = lane >> 4;
#pragma unroll
  for (int i = 0; i < 2; ++i)
#pragma unroll
    for (int j = 0; j < 4; ++j)
#pragma unroll
      for (int r = 0; r < 8; ++r) {
        int row = bm + wrow * 32 + i * 16 + r + half * 8;
        int col = bn + wcol * 64 + j * 16 + n;
        if (OUT_BF16)
          ((bf16_t*)Cv)[(long)row * N + col] = (bf16_t)acc[i][j][r];
        else
          ((float*)Cv)[(long)row * N + col] = acc[i][j][r];
      }
}

// ---------------------------------------------------------------------------
// Fused per-head RMSNorm + NeoX RoPE. One wave per (token, head) row of 128.
// ---------------------------------------------------------------------------
__global__ __launch_bounds__(128)
void rmsnorm_rope_kernel(const float* __restrict__ x,
                         const float* __restrict__ scale,
                         const int* __restrict__ positions,
                         bf16_t* __restrict__ out, int rows, int H) {
  int row  = blockIdx.x * 4 + (threadIdx.x >> 5);
  int lane = threadIdx.x & 31;
  if (row >= rows) return;
  int t = row / H;

  const float* px = x + (long)row * HEAD_DIM;
  float v0 = px[lane], v1 = px[lane + 32], v2 = px[lane + 64], v3 = px[lane + 96];
  float ss = v0 * v0 + v1 * v1 + v2 * v2 + v3 * v3;
#pragma unroll
  for (int m = 16; m >= 1; m >>= 1) ss += __shfl_xor(ss, m, 32);
  float inv = rsqrtf(ss * (1.0f / 128.0f) + 1e-6f);
  v0 *= inv * scale[lane];
  v1 *= inv * scale[lane + 32];
  v2 *= inv * scale[lane + 64];
  v3 *= inv * scale[lane + 96];

  float pos = (float)positions[t];
  float f0 = pos * __powf(1.0e6f, -((float)(2 * lane)) * (1.0f / 128.0f));
  float f1 = pos * __powf(1.0e6f, -((float)(2 * (lane + 32))) * (1.0f / 128.0f));
  float s0, c0, s1, c1;
  __sincosf(f0, &s0, &c0);
  __sincosf(f1, &s1, &c1);

  bf16_t* po = out + (long)row * HEAD_DIM;
  po[lane]      = (bf16_t)(v0 * c0 - v2 * s0);
  po[lane + 64] = (bf16_t)(v2 * c0 + v0 * s0);
  po[lane + 32] = (bf16_t)(v1 * c1 - v3 * s1);
  po[lane + 96] = (bf16_t)(v3 * c1 + v1 * s1);
}

// ---------------------------------------------------------------------------
// Flash-style causal GQA attention. Block = 128 threads (4 waves).
// Q staged once via TDM. K/V chunks double-buffered: TDM K(c+1) + register
// V(c+1) prefetch overlap with WMMA on chunk c; one barrier per chunk.
// ---------------------------------------------------------------------------
__global__ __launch_bounds__(128)
void attention_kernel(const bf16_t* __restrict__ Q, const bf16_t* __restrict__ Kc,
                      const bf16_t* __restrict__ Vc, bf16_t* __restrict__ O) {
  __shared__ bf16_t Qs[64][136];        // TDM pad: 64 DW + 4 DW -> stride 136
  __shared__ bf16_t Ks[2][32][136];
  __shared__ bf16_t Vst[2][128][40];    // transposed: [d][kv]
  __shared__ bf16_t Ps[4][16][40];

  const int head  = blockIdx.y;
  const int kvh   = head >> 3;
  const int qbase = blockIdx.x * 64;
  const int tid = threadIdx.x, lane = tid & 31, wave = tid >> 5;
  const int n = lane & 15, half = lane >> 4;
  const float sm_scale = 0.08838834764831845f;  // 1/sqrt(128)

  const int nchunks = (qbase + 64) / 32;  // causal trimming

  // ---- prologue: Q tile + chunk 0
  if (wave == 0) {
    tdm_load_2d(Q + (long)qbase * QDIM + head * HEAD_DIM, lds_offset_of(&Qs[0][0]),
                QDIM, T_SEQ, 128, 64, /*pad_icode=*/5, /*pad_acode=*/3);
    tdm_load_2d(Kc + kvh * HEAD_DIM, lds_offset_of(&Ks[0][0][0]),
                KVDIM, T_SEQ, 128, 32, 5, 3);
  }
  {
    v8bf vv[4];
#pragma unroll
    for (int it = 0; it < 4; ++it) {
      int u = tid + it * 128;
      int oc = (u >> 5) << 3, row = u & 31;
      vv[it] = *(const v8bf*)(Vc + (long)row * KVDIM + kvh * HEAD_DIM + oc);
    }
#pragma unroll
    for (int it = 0; it < 4; ++it) {
      int u = tid + it * 128;
      int oc = (u >> 5) << 3, row = u & 31;
#pragma unroll
      for (int e = 0; e < 8; ++e) Vst[0][oc + e][row] = vv[it][e];
    }
  }
  if (wave == 0) __builtin_amdgcn_s_wait_tensorcnt(0);
  __syncthreads();

  v16bf qfrag[4];
#pragma unroll
  for (int kk = 0; kk < 4; ++kk)
    qfrag[kk] = load_frag_a(&Qs[wave * 16][kk * 32], 136, lane);

  v8f o[8];
  float mrow[8], lrow[8];
#pragma unroll
  for (int dt = 0; dt < 8; ++dt) o[dt] = v8f_zero();
#pragma unroll
  for (int r = 0; r < 8; ++r) { mrow[r] = -1e30f; lrow[r] = 0.f; }

  int buf = 0;
  for (int c = 0; c < nchunks; ++c, buf ^= 1) {
    const int kbase = c * 32;
    const bool pref = (c + 1 < nchunks);
    // ---- prefetch chunk c+1 (TDM K -> LDS, V -> registers)
    v8bf vv[4];
    if (pref) {
      if (wave == 0)
        tdm_load_2d(Kc + (long)(kbase + 32) * KVDIM + kvh * HEAD_DIM,
                    lds_offset_of(&Ks[buf ^ 1][0][0]), KVDIM, T_SEQ, 128, 32, 5, 3);
#pragma unroll
      for (int it = 0; it < 4; ++it) {
        int u = tid + it * 128;
        int oc = (u >> 5) << 3, row = u & 31;
        vv[it] = *(const v8bf*)(Vc + (long)(kbase + 32 + row) * KVDIM +
                                kvh * HEAD_DIM + oc);
      }
    }

    // ---- S = Q @ K^T: preload all 8 B-fragments, then WMMA chain
    v16bf bk[8];
#pragma unroll
    for (int kk = 0; kk < 4; ++kk) {
      const bf16_t* p0 = &Ks[buf][n][kk * 32 + half * 16];
      const bf16_t* p1 = &Ks[buf][16 + n][kk * 32 + half * 16];
#pragma unroll
      for (int i = 0; i < 16; ++i) {
        bk[2 * kk][i]     = p0[i];
        bk[2 * kk + 1][i] = p1[i];
      }
    }
    v8f s0 = v8f_zero(), s1 = v8f_zero();
#pragma unroll
    for (int kk = 0; kk < 4; ++kk) {
      s0 = wmma_bf16(qfrag[kk], bk[2 * kk],     s0);
      s1 = wmma_bf16(qfrag[kk], bk[2 * kk + 1], s1);
    }

    // ---- mask + online softmax (C layout: row = r + 8*half, col = lane&15)
#pragma unroll
    for (int r = 0; r < 8; ++r) {
      int qrow = qbase + wave * 16 + r + half * 8;
      float a0 = s0[r] * sm_scale;
      float a1 = s1[r] * sm_scale;
      if (kbase + n > qrow)      a0 = -1e30f;
      if (kbase + 16 + n > qrow) a1 = -1e30f;
      float mloc = fmaxf(a0, a1);
#pragma unroll
      for (int msk = 8; msk >= 1; msk >>= 1)
        mloc = fmaxf(mloc, __shfl_xor(mloc, msk, 32));
      float mnew  = fmaxf(mrow[r], mloc);
      float alpha = __expf(mrow[r] - mnew);
      float p0 = __expf(a0 - mnew);
      float p1 = __expf(a1 - mnew);
      float psum = p0 + p1;
#pragma unroll
      for (int msk = 8; msk >= 1; msk >>= 1) psum += __shfl_xor(psum, msk, 32);
      lrow[r] = lrow[r] * alpha + psum;
      mrow[r] = mnew;
#pragma unroll
      for (int dt = 0; dt < 8; ++dt) o[dt][r] *= alpha;
      Ps[wave][r + half * 8][n]      = (bf16_t)p0;
      Ps[wave][r + half * 8][16 + n] = (bf16_t)p1;
    }
    // Ps is wave-private: wave-local DS drain is enough (no block barrier).
    asm volatile("s_wait_dscnt 0x0" ::: "memory");

    // ---- O += P @ V: preload all 8 V-fragments, then WMMA chain
    v16bf pfrag = load_frag_a(&Ps[wave][0][0], 40, lane);
    v16bf bvf[8];
#pragma unroll
    for (int dt = 0; dt < 8; ++dt) {
      const bf16_t* pv = &Vst[buf][dt * 16 + n][half * 16];
#pragma unroll
      for (int i = 0; i < 16; ++i) bvf[dt][i] = pv[i];
    }
#pragma unroll
    for (int dt = 0; dt < 8; ++dt) o[dt] = wmma_bf16(pfrag, bvf[dt], o[dt]);

    // ---- commit chunk c+1, release
    if (pref) {
#pragma unroll
      for (int it = 0; it < 4; ++it) {
        int u = tid + it * 128;
        int oc = (u >> 5) << 3, row = u & 31;
#pragma unroll
        for (int e = 0; e < 8; ++e) Vst[buf ^ 1][oc + e][row] = vv[it][e];
      }
      if (wave == 0) __builtin_amdgcn_s_wait_tensorcnt(0);
    }
    __syncthreads();
  }

  // ---- epilogue: O / l, write bf16
#pragma unroll
  for (int r = 0; r < 8; ++r) {
    float linv = 1.0f / lrow[r];
    int row = qbase + wave * 16 + r + half * 8;
#pragma unroll
    for (int dt = 0; dt < 8; ++dt)
      O[(long)row * QDIM + head * HEAD_DIM + dt * 16 + n] =
          (bf16_t)(o[dt][r] * linv);
  }
}

// ---------------------------------------------------------------------------
// Host-side orchestration
// ---------------------------------------------------------------------------
static inline size_t align256(size_t x) { return (x + 255) & ~(size_t)255; }

extern "C" void kernel_launch(void* const* d_in, const int* in_sizes, int n_in,
                              void* d_out, int out_size, void* d_ws, size_t ws_size,
                              hipStream_t stream) {
  (void)in_sizes; (void)n_in; (void)out_size; (void)ws_size;
  const int*   positions = (const int*)d_in[0];
  const float* hidden    = (const float*)d_in[1];
  const float* Wq        = (const float*)d_in[2];
  const float* Wk        = (const float*)d_in[3];
  const float* Wv        = (const float*)d_in[4];
  const float* Wo        = (const float*)d_in[5];
  const float* q_scale   = (const float*)d_in[6];
  const float* k_scale   = (const float*)d_in[7];
  float* out = (float*)d_out;

  char* p = (char*)d_ws;
  size_t off = 0;
  auto take = [&](size_t bytes) { char* r = p + off; off = align256(off + bytes); return (void*)r; };

  bf16_t* hid_b  = (bf16_t*)take((size_t)T_SEQ * HIDDEN * 2);
  bf16_t* wq_b   = (bf16_t*)take((size_t)HIDDEN * QDIM * 2);
  bf16_t* wk_b   = (bf16_t*)take((size_t)HIDDEN * KVDIM * 2);
  bf16_t* wv_b   = (bf16_t*)take((size_t)HIDDEN * KVDIM * 2);
  bf16_t* wo_b   = (bf16_t*)take((size_t)QDIM * HIDDEN * 2);
  float*  q_f    = (float*) take((size_t)T_SEQ * QDIM * 4);
  float*  k_f    = (float*) take((size_t)T_SEQ * KVDIM * 4);
  bf16_t* q_b    = (bf16_t*)take((size_t)T_SEQ * QDIM * 2);
  bf16_t* k_b    = (bf16_t*)take((size_t)T_SEQ * KVDIM * 2);
  bf16_t* v_b    = (bf16_t*)take((size_t)T_SEQ * KVDIM * 2);
  bf16_t* attn_b = (bf16_t*)take((size_t)T_SEQ * QDIM * 2);

  auto conv = [&](const float* src, bf16_t* dst, long cnt) {
    int blocks = (int)((cnt + 255) / 256);
    if (blocks > 2048) blocks = 2048;
    f32_to_bf16_kernel<<<blocks, 256, 0, stream>>>(src, dst, cnt);
  };
  conv(hidden, hid_b, (long)T_SEQ * HIDDEN);
  conv(Wq, wq_b, (long)HIDDEN * QDIM);
  conv(Wk, wk_b, (long)HIDDEN * KVDIM);
  conv(Wv, wv_b, (long)HIDDEN * KVDIM);
  conv(Wo, wo_b, (long)QDIM * HIDDEN);

  gemm_bf16_wmma<QDIM, HIDDEN, false>
      <<<dim3(QDIM / 128, T_SEQ / 128), 256, 0, stream>>>(hid_b, wq_b, q_f);
  gemm_bf16_wmma<KVDIM, HIDDEN, false>
      <<<dim3(KVDIM / 128, T_SEQ / 128), 256, 0, stream>>>(hid_b, wk_b, k_f);
  gemm_bf16_wmma<KVDIM, HIDDEN, true>   // V: bf16 out directly
      <<<dim3(KVDIM / 128, T_SEQ / 128), 256, 0, stream>>>(hid_b, wv_b, v_b);

  {
    int rows_q = T_SEQ * NUM_HEADS;
    rmsnorm_rope_kernel<<<rows_q / 4, 128, 0, stream>>>(q_f, q_scale, positions,
                                                        q_b, rows_q, NUM_HEADS);
    int rows_k = T_SEQ * NUM_KVH;
    rmsnorm_rope_kernel<<<rows_k / 4, 128, 0, stream>>>(k_f, k_scale, positions,
                                                        k_b, rows_k, NUM_KVH);
  }

  attention_kernel<<<dim3(T_SEQ / 64, NUM_HEADS), 128, 0, stream>>>(q_b, k_b,
                                                                    v_b, attn_b);

  gemm_bf16_wmma<HIDDEN, QDIM, false>
      <<<dim3(HIDDEN / 128, T_SEQ / 128), 256, 0, stream>>>(attn_b, wo_b, out);
}
